// MultiHeadSelfAttention_7516192768209
// MI455X (gfx1250) — matrix operand order repaired and checked by reading, not compiled, not run on hardware
//
#include <hip/hip_runtime.h>

typedef __attribute__((ext_vector_type(16))) __bf16 v16bf;
typedef __attribute__((ext_vector_type(8)))  float  v8f;
typedef __attribute__((ext_vector_type(4)))  int    v4i;
typedef __attribute__((ext_vector_type(4)))  unsigned int v4u;

#define EMBED 1024
#define NHEAD 16
#define HDIM  64
#define SEQ   2048
#define BATCH 2
#define N_QKV (3 * EMBED)

#if defined(__AMDGCN__) && __has_builtin(__builtin_amdgcn_global_load_async_to_lds_b128)
#define USE_ASYNC_LDS 1
// pointer-to-int4 in global (AS1) / LDS (AS3) address spaces
typedef __attribute__((address_space(1))) v4i* as1_v4i_ptr;
typedef __attribute__((address_space(3))) v4i* as3_v4i_ptr;
#else
#define USE_ASYNC_LDS 0
#endif

__device__ __forceinline__ __bf16 f2bf(float f) { return (__bf16)f; }

__device__ __forceinline__ v8f wmma_bf16(v16bf a, v16bf b, v8f c) {
  // (neg_a, A, neg_b, B, c_mod, C, reuse_a, reuse_b)
  return __builtin_amdgcn_wmma_f32_16x16x32_bf16(false, a, false, b, (short)0, c,
                                                 false, false);
}

// A fragment: 16x32 bf16 tile at p (row-major, leading dim ld).
// lane: m = lane%16, half = lane/16; element i: k = half*16 + i
__device__ __forceinline__ v16bf load_a_frag(const __bf16* p, int ld) {
  const int lane = threadIdx.x & 31;
  const int m = lane & 15, half = lane >> 4;
  v16bf a;
#pragma unroll
  for (int i = 0; i < 16; ++i)
    a[i] = p[m * ld + half * 16 + i];
  return a;
}

// B fragment: 32x16 where storage is column-contiguous: p[n*ld + k].
// lane: n = lane%16, half = lane/16; element i: k = half*16 + i
__device__ __forceinline__ v16bf load_b_frag(const __bf16* p, int ld) {
  const int lane = threadIdx.x & 31;
  const int n = lane & 15, half = lane >> 4;
  v16bf b;
#pragma unroll
  for (int i = 0; i < 16; ++i)
    b[i] = p[n * ld + half * 16 + i];
  return b;
}

// 16-byte global -> LDS copy: async DMA on gfx1250, sync fallback otherwise.
__device__ __forceinline__ void async_copy16(const __bf16* g, __bf16* l) {
#if USE_ASYNC_LDS
  __builtin_amdgcn_global_load_async_to_lds_b128((as1_v4i_ptr)(g),
                                                 (as3_v4i_ptr)(l), 0, 0);
#else
  *(v4u*)l = *(const v4u*)g;
#endif
}

__device__ __forceinline__ void async_wait0() {
#if USE_ASYNC_LDS
#if __has_builtin(__builtin_amdgcn_s_wait_asynccnt)
  __builtin_amdgcn_s_wait_asynccnt(0);
#else
  asm volatile("s_wait_asynccnt 0x0" ::: "memory");
#endif
#endif
}

__device__ __forceinline__ void async_wait8() {
#if USE_ASYNC_LDS
#if __has_builtin(__builtin_amdgcn_s_wait_asynccnt)
  __builtin_amdgcn_s_wait_asynccnt(8);
#else
  asm volatile("s_wait_asynccnt 0x8" ::: "memory");
#endif
#endif
}

// ---------------------------------------------------------------------------
// Kernel 1: qkv = x @ W_qkv + b_qkv ; scatter to bf16 Q[B,H,S,Hd] (prescaled
// by 1/sqrt(Hd)), K[B,H,S,Hd], and V TRANSPOSED per head [B,H,Hd,S] so the
// attention kernel can stage V tiles with straight async row copies.
// Block tile 128x128, 8 waves, each wave 32(M)x64(N), KC=64.
// ---------------------------------------------------------------------------
__global__ __launch_bounds__(256) void qkv_gemm_kernel(
    const float* __restrict__ x, const float* __restrict__ W,
    const float* __restrict__ bias, __bf16* __restrict__ Qb,
    __bf16* __restrict__ Kb, __bf16* __restrict__ Vtb) {
  __shared__ alignas(16) __bf16 As[128][64];  // [m][k]
  __shared__ alignas(16) __bf16 Bs[128][64];  // [n][k]
  const int tid = threadIdx.x;
  const int lane = tid & 31, wave = tid >> 5;
  const int wm = wave >> 1, wn = wave & 1;
  const int m0 = blockIdx.y * 128, n0 = blockIdx.x * 128;

  v8f acc[2][4] = {};

  for (int ko = 0; ko < EMBED; ko += 64) {
#pragma unroll
    for (int i = tid; i < 128 * 64; i += 256) {
      int r = i >> 6, c = i & 63;
      As[r][c] = f2bf(x[(m0 + r) * EMBED + ko + c]);
    }
#pragma unroll
    for (int i = tid; i < 64 * 128; i += 256) {
      int r = i >> 7, c = i & 127;  // r=k, c=n (coalesced global read)
      Bs[c][r] = f2bf(W[(ko + r) * N_QKV + n0 + c]);
    }
    if (ko + 64 < EMBED) {  // prefetch next K-slab (emits global_prefetch_b8)
      __builtin_prefetch(&x[(m0 + (tid >> 1)) * EMBED + ko + 64 + (tid & 1) * 32], 0, 1);
      __builtin_prefetch(&W[(ko + 64 + (tid >> 2)) * N_QKV + n0 + (tid & 3) * 32], 0, 1);
    }
    __syncthreads();

#pragma unroll
    for (int ks = 0; ks < 2; ++ks) {
      v16bf a0 = load_a_frag(&As[wm * 32][ks * 32], 64);
      v16bf a1 = load_a_frag(&As[wm * 32 + 16][ks * 32], 64);
#pragma unroll
      for (int ni = 0; ni < 4; ++ni) {
        v16bf bfr = load_b_frag(&Bs[wn * 64 + ni * 16][ks * 32], 64);
        acc[0][ni] = wmma_bf16(a0, bfr, acc[0][ni]);
        acc[1][ni] = wmma_bf16(a1, bfr, acc[1][ni]);
      }
    }
    __syncthreads();
  }

  const int half = lane >> 4, nl = lane & 15;
#pragma unroll
  for (int mi = 0; mi < 2; ++mi)
#pragma unroll
    for (int ni = 0; ni < 4; ++ni)
#pragma unroll
      for (int r = 0; r < 8; ++r) {
        int m = m0 + wm * 32 + mi * 16 + r + half * 8;
        int n = n0 + wn * 64 + ni * 16 + nl;
        float v = acc[mi][ni][r] + bias[n];
        int comp = n >> 10;  // 0=q 1=k 2=v
        int rem = n & (EMBED - 1);
        int hh = rem >> 6, hd = rem & 63;
        int bb = m >> 11, ss = m & (SEQ - 1);
        if (comp == 0)
          Qb[(((bb << 4) + hh) * SEQ + ss) * HDIM + hd] = f2bf(v * 0.125f);
        else if (comp == 1)
          Kb[(((bb << 4) + hh) * SEQ + ss) * HDIM + hd] = f2bf(v);
        else
          Vtb[(((bb << 4) + hh) * HDIM + hd) * SEQ + ss] = f2bf(v);
      }
}

// Stage one 64x64 K tile ([key][hd], contiguous) and one 64x64 V^T tile
// ([hd][key], 128B rows) into LDS. 8 async issues per thread.
__device__ __forceinline__ void stage_kv(const __bf16* K, const __bf16* Vt,
                                         __bf16* ks, __bf16* vts, int k0,
                                         int tid) {
#pragma unroll
  for (int j = 0; j < 4; ++j) {
    int c = tid + j * 128;  // 16B chunk id, 512 chunks = 8KB
    async_copy16(K + k0 * HDIM + c * 8, ks + c * 8);
  }
#pragma unroll
  for (int j = 0; j < 4; ++j) {
    int c = tid + j * 128;
    int row = c >> 3, cc = c & 7;  // 8 x 16B chunks per 128B row
    async_copy16(Vt + row * SEQ + k0 + cc * 8, vts + row * 64 + cc * 8);
  }
}

// ---------------------------------------------------------------------------
// Kernel 2: causal flash attention. One workgroup per (b, h, 64-row q tile),
// 4 waves x 16 q-rows. Double-buffered async K/V staging, online softmax in
// f32, QK^T and PV on bf16 WMMA.
// ---------------------------------------------------------------------------
__global__ __launch_bounds__(128) void attn_kernel(
    const __bf16* __restrict__ Qb, const __bf16* __restrict__ Kb,
    const __bf16* __restrict__ Vtb, __bf16* __restrict__ ctx) {
  __shared__ alignas(16) __bf16 Qs[64][64];        // [q][hd]
  __shared__ alignas(16) __bf16 Ks[2][64][64];     // [key][hd]
  __shared__ alignas(16) __bf16 Vts[2][64][64];    // [hd][key]
  __shared__ alignas(16) __bf16 Ps[4][16][64];     // wave-private P staging

  const int qt = blockIdx.x, h = blockIdx.y, b = blockIdx.z;
  const int tid = threadIdx.x, lane = tid & 31, wave = tid >> 5;
  const int half = lane >> 4, nl = lane & 15;
  const __bf16* Q = Qb + ((b * NHEAD + h) * SEQ) * HDIM;
  const __bf16* K = Kb + ((b * NHEAD + h) * SEQ) * HDIM;
  const __bf16* Vt = Vtb + ((b * NHEAD + h) * HDIM) * SEQ;
  const int q0 = qt * 64;

  // stage Q tile (contiguous 8KB) + first K/V tiles, then drain
#pragma unroll
  for (int j = 0; j < 4; ++j) {
    int c = tid + j * 128;
    async_copy16(Q + q0 * HDIM + c * 8, &Qs[0][0] + c * 8);
  }
  stage_kv(K, Vt, &Ks[0][0][0], &Vts[0][0][0], 0, tid);
  async_wait0();
  __syncthreads();

  v16bf qf0 = load_a_frag(&Qs[wave * 16][0], 64);
  v16bf qf1 = load_a_frag(&Qs[wave * 16][32], 64);

  float Mrow[8], Lrow[8];
  v8f O[4] = {};
#pragma unroll
  for (int r = 0; r < 8; ++r) {
    Mrow[r] = -1e30f;
    Lrow[r] = 0.f;
  }

  for (int kt = 0; kt <= qt; ++kt) {
    const int k0 = kt * 64;
    const int cur = kt & 1;
    __syncthreads();  // all waves done reading buf[1-cur] from iter kt-1
    if (kt < qt) {    // prefetch next tile into the other buffer
      stage_kv(K, Vt, &Ks[1 - cur][0][0], &Vts[1 - cur][0][0], k0 + 64, tid);
      async_wait8();  // in-order: retires current tile, next stays in flight
    } else {
      async_wait0();
    }
    __syncthreads();

    // S = Q @ K^T  (per wave: 16 x 64 scores)
    v8f sc[4];
#pragma unroll
    for (int ni = 0; ni < 4; ++ni) {
      v8f z = {};
      z = wmma_bf16(qf0, load_b_frag(&Ks[cur][ni * 16][0], 64), z);
      z = wmma_bf16(qf1, load_b_frag(&Ks[cur][ni * 16][32], 64), z);
      sc[ni] = z;
    }

    // causal mask + online softmax update
#pragma unroll
    for (int r = 0; r < 8; ++r) {
      const int row = q0 + wave * 16 + r + half * 8;
      float mx = -1e30f;
#pragma unroll
      for (int ni = 0; ni < 4; ++ni) {
        int col = k0 + ni * 16 + nl;
        float s = (col <= row) ? sc[ni][r] : -1e30f;
        sc[ni][r] = s;
        mx = fmaxf(mx, s);
      }
#pragma unroll
      for (int d = 1; d < 16; d <<= 1)
        mx = fmaxf(mx, __shfl_xor(mx, d, 32));
      float mnew = fmaxf(Mrow[r], mx);
      float alpha = __expf(Mrow[r] - mnew);
      float rsum = 0.f;
#pragma unroll
      for (int ni = 0; ni < 4; ++ni) {
        float p = __expf(sc[ni][r] - mnew);
        sc[ni][r] = p;
        rsum += p;
      }
#pragma unroll
      for (int d = 1; d < 16; d <<= 1) rsum += __shfl_xor(rsum, d, 32);
      Mrow[r] = mnew;
      Lrow[r] = Lrow[r] * alpha + rsum;
#pragma unroll
      for (int ni = 0; ni < 4; ++ni) O[ni][r] = O[ni][r] * alpha;
    }

    // C-layout -> A-layout via wave-private LDS staging (in-wave DS ordering)
#pragma unroll
    for (int ni = 0; ni < 4; ++ni)
#pragma unroll
      for (int r = 0; r < 8; ++r)
        Ps[wave][r + half * 8][ni * 16 + nl] = f2bf(sc[ni][r]);

    // O += P @ V
#pragma unroll
    for (int ks = 0; ks < 2; ++ks) {
      v16bf pf = load_a_frag(&Ps[wave][0][ks * 32], 64);
#pragma unroll
      for (int ni = 0; ni < 4; ++ni)
        O[ni] = wmma_bf16(pf, load_b_frag(&Vts[cur][ni * 16][ks * 32], 64),
                          O[ni]);
    }
  }

  // normalize and write context in [B,S,D] layout for the out-proj GEMM
#pragma unroll
  for (int ni = 0; ni < 4; ++ni)
#pragma unroll
    for (int r = 0; r < 8; ++r) {
      int s = q0 + wave * 16 + r + half * 8;
      int hd = ni * 16 + nl;
      float o = O[ni][r] / Lrow[r];
      ctx[(b * SEQ + s) * EMBED + h * HDIM + hd] = f2bf(o);
    }
}

// ---------------------------------------------------------------------------
// Kernel 3: out = ctx @ W_out + b_out (f32 output). Same shape as kernel 1.
// ---------------------------------------------------------------------------
__global__ __launch_bounds__(256) void out_gemm_kernel(
    const __bf16* __restrict__ ctx, const float* __restrict__ W,
    const float* __restrict__ bias, float* __restrict__ out) {
  __shared__ alignas(16) __bf16 As[128][64];
  __shared__ alignas(16) __bf16 Bs[128][64];
  const int tid = threadIdx.x;
  const int lane = tid & 31, wave = tid >> 5;
  const int wm = wave >> 1, wn = wave & 1;
  const int m0 = blockIdx.y * 128, n0 = blockIdx.x * 128;

  v8f acc[2][4] = {};

  for (int ko = 0; ko < EMBED; ko += 64) {
#pragma unroll
    for (int i = tid; i < 128 * 64; i += 256) {
      int r = i >> 6, c = i & 63;
      As[r][c] = ctx[(m0 + r) * EMBED + ko + c];
    }
#pragma unroll
    for (int i = tid; i < 64 * 128; i += 256) {
      int r = i >> 7, c = i & 127;
      Bs[c][r] = f2bf(W[(ko + r) * EMBED + n0 + c]);
    }
    if (ko + 64 < EMBED) {
      __builtin_prefetch(&ctx[(m0 + (tid >> 1)) * EMBED + ko + 64 + (tid & 1) * 32], 0, 1);
      __builtin_prefetch(&W[(ko + 64 + (tid >> 2)) * EMBED + n0 + (tid & 3) * 32], 0, 1);
    }
    __syncthreads();

#pragma unroll
    for (int ks = 0; ks < 2; ++ks) {
      v16bf a0 = load_a_frag(&As[wm * 32][ks * 32], 64);
      v16bf a1 = load_a_frag(&As[wm * 32 + 16][ks * 32], 64);
#pragma unroll
      for (int ni = 0; ni < 4; ++ni) {
        v16bf bfr = load_b_frag(&Bs[wn * 64 + ni * 16][ks * 32], 64);
        acc[0][ni] = wmma_bf16(a0, bfr, acc[0][ni]);
        acc[1][ni] = wmma_bf16(a1, bfr, acc[1][ni]);
      }
    }
    __syncthreads();
  }

  const int half = lane >> 4, nl = lane & 15;
#pragma unroll
  for (int mi = 0; mi < 2; ++mi)
#pragma unroll
    for (int ni = 0; ni < 4; ++ni)
#pragma unroll
      for (int r = 0; r < 8; ++r) {
        int m = m0 + wm * 32 + mi * 16 + r + half * 8;
        int n = n0 + wn * 64 + ni * 16 + nl;
        out[m * EMBED + n] = acc[mi][ni][r] + bias[n];
      }
}

// ---------------------------------------------------------------------------
extern "C" void kernel_launch(void* const* d_in, const int* in_sizes, int n_in,
                              void* d_out, int out_size, void* d_ws,
                              size_t ws_size, hipStream_t stream) {
  (void)in_sizes; (void)n_in; (void)out_size; (void)ws_size;
  const float* x = (const float*)d_in[0];
  const float* Wqkv = (const float*)d_in[1];
  const float* bqkv = (const float*)d_in[2];
  const float* Wout = (const float*)d_in[3];
  const float* bout = (const float*)d_in[4];
  float* out = (float*)d_out;

  const size_t QKV_ELEMS = (size_t)BATCH * NHEAD * SEQ * HDIM;  // 4M each
  __bf16* Qb = (__bf16*)d_ws;
  __bf16* Kb = Qb + QKV_ELEMS;
  __bf16* Vtb = Kb + QKV_ELEMS;       // [B,H,Hd,S]
  __bf16* ctx = Vtb + QKV_ELEMS;      // [B,S,D]

  qkv_gemm_kernel<<<dim3(N_QKV / 128, (BATCH * SEQ) / 128), 256, 0, stream>>>(
      x, Wqkv, bqkv, Qb, Kb, Vtb);
  attn_kernel<<<dim3(SEQ / 64, NHEAD, BATCH), 128, 0, stream>>>(Qb, Kb, Vtb,
                                                                ctx);
  out_gemm_kernel<<<dim3(EMBED / 128, (BATCH * SEQ) / 128), 256, 0, stream>>>(
      ctx, Wout, bout, out);
}
